// AttentionCritic_66271345377613
// MI455X (gfx1250) — matrix operand order repaired and checked by reading, not compiled, not run on hardware
//
#include <hip/hip_runtime.h>

// ---------------------------------------------------------------------------
// AttentionCritic fused kernel for gfx1250 (MI455X, wave32, WMMA bf16)
// Shapes: A=8 agents, B=32768 batch, D=152(obs128+skill16+param8, pad->160),
//         H=256, NH=4, HD=64.
// Strategy: per-workgroup tile of 16 batch rows; entire pipeline fused,
// all intermediates live in LDS; only weights (bf16, transposed [N][Kpad])
// are staged in the workspace by a small prep kernel.
// ---------------------------------------------------------------------------

#define A_N   8
#define OBS_D 128
#define NS_D  16
#define PD_D  8
#define D_PAD 160
#define H_DIM 256
#define BT    16

typedef __attribute__((ext_vector_type(16))) __bf16 v16bf;
typedef __attribute__((ext_vector_type(8)))  float  v8f;

union ABFrag { v16bf v; uint4 q[2]; };
union CDFrag { v8f  v; float f[8]; };

__device__ __forceinline__ unsigned short f2bf(float f) {
  unsigned u = __float_as_uint(f);
  u += 0x7FFFu + ((u >> 16) & 1u);          // round-to-nearest-even
  return (unsigned short)(u >> 16);
}
__device__ __forceinline__ float bf2f(unsigned short h) {
  return __uint_as_float(((unsigned)h) << 16);
}

// ---- WMMA micro-GEMM: one 16-row M-tile x NT N-tiles, A from LDS (bf16,
// row-major, stride aStride), B from global bf16 transposed [n][kPad].
template <int NT>
__device__ __forceinline__ void gemm_rows16(
    const unsigned short* __restrict__ As, int aStride, int srcRow0,
    const unsigned short* __restrict__ Bt, int kPad, int nTile0,
    int kSteps, CDFrag acc[NT]) {
  const int lane = threadIdx.x & 31;
  const int m    = lane & 15;
  const int half = lane >> 4;
  const int nIn  = lane & 15;
  for (int ks = 0; ks < kSteps; ++ks) {
    const int kb = ks * 32;
    ABFrag a;
    const unsigned short* ap = As + (srcRow0 + m) * aStride + kb + half * 8;
    a.q[0] = *(const uint4*)(ap);        // K = half*8 + 0..7
    a.q[1] = *(const uint4*)(ap + 16);   // K = 16 + half*8 + 0..7
#pragma unroll
    for (int t = 0; t < NT; ++t) {
      ABFrag b;
      const unsigned short* bp =
          Bt + (size_t)((nTile0 + t) * 16 + nIn) * kPad + kb + half * 8;
      b.q[0] = *(const uint4*)(bp);      // K = half*8 + 0..7
      b.q[1] = *(const uint4*)(bp + 16); // K = 16 + half*8 + 0..7
      acc[t].v = __builtin_amdgcn_wmma_f32_16x16x32_bf16(
          false, a.v, false, b.v, (short)0, acc[t].v, false, false);
    }
  }
}

template <int NT>
__device__ __forceinline__ void epilogue_bf16(
    unsigned short* dst, int stride, int dstRow0, int nTile0,
    const CDFrag acc[NT], const float* __restrict__ bias, bool relu) {
  const int lane = threadIdx.x & 31;
  const int half = lane >> 4;
  const int nIn  = lane & 15;
#pragma unroll
  for (int t = 0; t < NT; ++t) {
    const int n  = (nTile0 + t) * 16 + nIn;
    const float bv = bias[n];
#pragma unroll
    for (int i = 0; i < 8; ++i) {
      float v = acc[t].f[i] + bv;
      if (relu) v = fmaxf(v, 0.0f);
      dst[(dstRow0 + i + half * 8) * stride + n] = f2bf(v);
    }
  }
}

template <int NT>
__device__ __forceinline__ void epilogue_f32(
    float* dst, int stride, int dstRow0, int nTile0,
    const CDFrag acc[NT], const float* __restrict__ bias, bool relu) {
  const int lane = threadIdx.x & 31;
  const int half = lane >> 4;
  const int nIn  = lane & 15;
#pragma unroll
  for (int t = 0; t < NT; ++t) {
    const int n  = (nTile0 + t) * 16 + nIn;
    const float bv = bias[n];
#pragma unroll
    for (int i = 0; i < 8; ++i) {
      float v = acc[t].f[i] + bv;
      if (relu) v = fmaxf(v, 0.0f);
      dst[(dstRow0 + i + half * 8) * stride + n] = v;
    }
  }
}

// ---- Weight prep: f32 [K,N] -> bf16 transposed [N,Kpad] in workspace -------
__device__ __forceinline__ void tconv(unsigned short* dst, const float* src,
                                      int K, int N, int Kpad, int tid, int nt) {
  for (int idx = tid; idx < N * Kpad; idx += nt) {
    const int n = idx / Kpad;
    const int k = idx - n * Kpad;
    dst[idx] = (k < K) ? f2bf(src[k * N + n]) : (unsigned short)0;
  }
}

// workspace offsets in ushort elements
#define OFF_WE1T 0                       // 256*160
#define OFF_WE2T (OFF_WE1T + 256 * 160)  // 256*256
#define OFF_WQT  (OFF_WE2T + 256 * 256)
#define OFF_WKT  (OFF_WQT  + 256 * 256)
#define OFF_WVT  (OFF_WKT  + 256 * 256)
#define OFF_WOT  (OFF_WVT  + 256 * 256)
#define OFF_WH1T (OFF_WOT  + 256 * 256)  // 256*512

__global__ void ac_prep_weights(const float* We1, const float* We2,
                                const float* Wq, const float* Wk,
                                const float* Wv, const float* Wo,
                                const float* Wh1, unsigned short* ws) {
  const int tid = blockIdx.x * blockDim.x + threadIdx.x;
  const int nt  = gridDim.x * blockDim.x;
  tconv(ws + OFF_WE1T, We1, 152, 256, 160, tid, nt);
  tconv(ws + OFF_WE2T, We2, 256, 256, 256, tid, nt);
  tconv(ws + OFF_WQT,  Wq,  256, 256, 256, tid, nt);
  tconv(ws + OFF_WKT,  Wk,  256, 256, 256, tid, nt);
  tconv(ws + OFF_WVT,  Wv,  256, 256, 256, tid, nt);
  tconv(ws + OFF_WOT,  Wo,  256, 256, 256, tid, nt);
  tconv(ws + OFF_WH1T, Wh1, 512, 256, 512, tid, nt);
}

// ---- Fused main kernel -----------------------------------------------------
struct FArgs {
  const float *obs, *skill, *prm;
  const unsigned short *we1t, *we2t, *wqt, *wkt, *wvt, *wot, *wh1t;
  const float *be1, *be2, *bq, *bk, *bv, *bo, *lng, *lnb, *bh1, *wh2, *bh2;
  const int* agent_idx;
  float* out;
  int B;
};

// LDS byte layout (160 KB total, regions reused across phases):
//   e1   [0,      65536)  : enc1 tile  128x256 bf16   (P1->P2)
//   xt   [65536, 106496)  : x tile     128x160 bf16   (P0->P1)
//   encs [65536, 131072)  : enc tile   128x256 bf16   (P2->P5)
//   qf   [131072,147456)  : q proj     16x256  f32    (P3b->P4a)
//   aw   [147456,149504)  : softmax    16x4x8  f32    (P4a->P4b)
//   ksm  [0,      65536)  : K proj     128x256 bf16   (P3a->P4a) [over e1]
//   vsm  [0,      65536)  : V proj     128x256 bf16   (P3c->P4b) [over ksm]
//   ctxs [131072,139264)  : context    16x256  bf16   (P4b->P5)  [over qf]
//   yt   [147456,163840)  : attn_out+res 16x256 f32   (P5)       [over aw]
//   comb [0,      16384)  : [q_enc,ln] 16x512  bf16   (P5->P6)   [over vsm]
//   hs   [16384,  32768)  : head1 out  16x256  f32    (P6)       [over vsm]

__global__ __launch_bounds__(256, 1) void ac_fused(FArgs p) {
  __shared__ __align__(16) unsigned char smem[163840];
  unsigned short* e1   = (unsigned short*)(smem + 0);
  unsigned short* xt   = (unsigned short*)(smem + 65536);
  unsigned short* encs = (unsigned short*)(smem + 65536);
  float*          qf   = (float*)(smem + 131072);
  float*          aw   = (float*)(smem + 147456);
  unsigned short* ksm  = (unsigned short*)(smem + 0);
  unsigned short* vsm  = (unsigned short*)(smem + 0);
  unsigned short* ctxs = (unsigned short*)(smem + 131072);
  float*          yt   = (float*)(smem + 147456);
  unsigned short* comb = (unsigned short*)(smem + 0);
  float*          hs   = (float*)(smem + 16384);

  const int tid  = threadIdx.x;
  const int wav  = tid >> 5;
  const int b0   = blockIdx.x * BT;
  const int aq   = *p.agent_idx;
  const int B    = p.B;

  // ---- P0: stage x = concat(obs, skill, params) as bf16, pad K to 160 -----
  for (int idx = tid; idx < 128 * 32; idx += 256) {   // obs: 128 rows x 32 f4
    const int row = idx >> 5, c4 = idx & 31;
    const int ag = row >> 4, r = row & 15;
    const float4 v = *(const float4*)(p.obs + ((size_t)ag * B + b0 + r) * OBS_D + c4 * 4);
    uint2 pk;
    pk.x = (unsigned)f2bf(v.x) | ((unsigned)f2bf(v.y) << 16);
    pk.y = (unsigned)f2bf(v.z) | ((unsigned)f2bf(v.w) << 16);
    *(uint2*)(xt + row * D_PAD + c4 * 4) = pk;
  }
  for (int idx = tid; idx < 128 * 4; idx += 256) {    // skill: 128 rows x 4 f4
    const int row = idx >> 2, c4 = idx & 3;
    const int ag = row >> 4, r = row & 15;
    const float4 v = *(const float4*)(p.skill + ((size_t)ag * B + b0 + r) * NS_D + c4 * 4);
    uint2 pk;
    pk.x = (unsigned)f2bf(v.x) | ((unsigned)f2bf(v.y) << 16);
    pk.y = (unsigned)f2bf(v.z) | ((unsigned)f2bf(v.w) << 16);
    *(uint2*)(xt + row * D_PAD + OBS_D + c4 * 4) = pk;
  }
  for (int idx = tid; idx < 128 * 2; idx += 256) {    // params: 128 rows x 2 f4
    const int row = idx >> 1, c4 = idx & 1;
    const int ag = row >> 4, r = row & 15;
    const float4 v = *(const float4*)(p.prm + ((size_t)ag * B + b0 + r) * PD_D + c4 * 4);
    uint2 pk;
    pk.x = (unsigned)f2bf(v.x) | ((unsigned)f2bf(v.y) << 16);
    pk.y = (unsigned)f2bf(v.z) | ((unsigned)f2bf(v.w) << 16);
    *(uint2*)(xt + row * D_PAD + OBS_D + NS_D + c4 * 4) = pk;
  }
  for (int row = tid; row < 128; row += 256) {        // zero pad cols 152..159
    uint4 z; z.x = z.y = z.z = z.w = 0u;
    *(uint4*)(xt + row * D_PAD + 152) = z;
  }
  __syncthreads();

  // ---- P1: enc1 = relu(x @ We1 + be1)   [128x160]x[160x256] ---------------
  {
    CDFrag acc[16];
#pragma unroll
    for (int t = 0; t < 16; ++t) acc[t].v = v8f{0, 0, 0, 0, 0, 0, 0, 0};
    gemm_rows16<16>(xt, D_PAD, wav * 16, p.we1t, 160, 0, 5, acc);
    epilogue_bf16<16>(e1, H_DIM, wav * 16, 0, acc, p.be1, true);
  }
  __syncthreads();

  // ---- P2: enc = enc1 @ We2 + be2   [128x256]x[256x256] -------------------
  {
    CDFrag acc[16];
#pragma unroll
    for (int t = 0; t < 16; ++t) acc[t].v = v8f{0, 0, 0, 0, 0, 0, 0, 0};
    gemm_rows16<16>(e1, H_DIM, wav * 16, p.we2t, 256, 0, 8, acc);
    epilogue_bf16<16>(encs, H_DIM, wav * 16, 0, acc, p.be2, false);
  }
  __syncthreads();

  // ---- P3a: K = enc @ Wk + bk  (all 8 agents) -----------------------------
  {
    CDFrag acc[16];
#pragma unroll
    for (int t = 0; t < 16; ++t) acc[t].v = v8f{0, 0, 0, 0, 0, 0, 0, 0};
    gemm_rows16<16>(encs, H_DIM, wav * 16, p.wkt, 256, 0, 8, acc);
    epilogue_bf16<16>(ksm, H_DIM, wav * 16, 0, acc, p.bk, false);
  }
  // ---- P3b: q = enc[aq] @ Wq + bq  (f32, 2 N-tiles per wave) --------------
  {
    CDFrag acc[2];
#pragma unroll
    for (int t = 0; t < 2; ++t) acc[t].v = v8f{0, 0, 0, 0, 0, 0, 0, 0};
    gemm_rows16<2>(encs, H_DIM, aq * 16, p.wqt, 256, wav * 2, 8, acc);
    epilogue_f32<2>(qf, H_DIM, 0, wav * 2, acc, p.bq, false);
  }
  __syncthreads();

  // ---- P4a: scores + softmax over 8 agents (one thread per (b,h)) ---------
  if (tid < 64) {
    const int r = tid >> 2, h = tid & 3;
    const float* q = qf + r * H_DIM + h * 64;
    float s[A_N];
    float mx = -1e30f;
#pragma unroll
    for (int a = 0; a < A_N; ++a) {
      const unsigned short* kk = ksm + (a * 16 + r) * H_DIM + h * 64;
      float acc = 0.0f;
      for (int d = 0; d < 64; ++d) acc += q[d] * bf2f(kk[d]);
      s[a] = acc * 0.125f;               // 1/sqrt(64)
      mx = fmaxf(mx, s[a]);
    }
    float sum = 0.0f;
#pragma unroll
    for (int a = 0; a < A_N; ++a) { s[a] = __expf(s[a] - mx); sum += s[a]; }
    const float inv = 1.0f / sum;
#pragma unroll
    for (int a = 0; a < A_N; ++a) aw[(r * 4 + h) * A_N + a] = s[a] * inv;
  }
  __syncthreads();

  // ---- P3c: V = enc @ Wv + bv  (overwrites K region) ----------------------
  {
    CDFrag acc[16];
#pragma unroll
    for (int t = 0; t < 16; ++t) acc[t].v = v8f{0, 0, 0, 0, 0, 0, 0, 0};
    gemm_rows16<16>(encs, H_DIM, wav * 16, p.wvt, 256, 0, 8, acc);
    epilogue_bf16<16>(vsm, H_DIM, wav * 16, 0, acc, p.bv, false);
  }
  __syncthreads();

  // ---- P4b: ctx = attn @ V ------------------------------------------------
  if (tid < 64) {
    const int r = tid >> 2, h = tid & 3;
    const float* w = aw + (r * 4 + h) * A_N;
    for (int d = 0; d < 64; ++d) {
      float acc = 0.0f;
#pragma unroll
      for (int a = 0; a < A_N; ++a)
        acc += w[a] * bf2f(vsm[(a * 16 + r) * H_DIM + h * 64 + d]);
      ctxs[r * H_DIM + h * 64 + d] = f2bf(acc);
    }
  }
  __syncthreads();

  // ---- P5: attn_out = ctx @ Wo + bo  (2 N-tiles/wave, M=16) ---------------
  {
    CDFrag acc[2];
#pragma unroll
    for (int t = 0; t < 2; ++t) acc[t].v = v8f{0, 0, 0, 0, 0, 0, 0, 0};
    gemm_rows16<2>(ctxs, H_DIM, 0, p.wot, 256, wav * 2, 8, acc);
    epilogue_f32<2>(yt, H_DIM, 0, wav * 2, acc, p.bo, false);
  }
  __syncthreads();

  // ---- P5b: residual + LayerNorm; build combined=[q_enc, ln] bf16 ---------
  if (tid < BT) {
    const int r = tid;
    const unsigned short* encrow = encs + (aq * 16 + r) * H_DIM;
    float mu = 0.0f;
    for (int n = 0; n < H_DIM; ++n) {
      const float yv = yt[r * H_DIM + n] + bf2f(encrow[n]);
      yt[r * H_DIM + n] = yv;
      mu += yv;
    }
    mu *= (1.0f / H_DIM);
    float var = 0.0f;
    for (int n = 0; n < H_DIM; ++n) {
      const float d = yt[r * H_DIM + n] - mu;
      var += d * d;
    }
    const float rstd = rsqrtf(var * (1.0f / H_DIM) + 1e-5f);
    for (int n = 0; n < H_DIM; ++n) {
      const float lnv = (yt[r * H_DIM + n] - mu) * rstd * p.lng[n] + p.lnb[n];
      comb[r * 512 + n]         = encrow[n];     // q_enc half (already bf16)
      comb[r * 512 + H_DIM + n] = f2bf(lnv);     // ln half
    }
  }
  __syncthreads();

  // ---- P6: h = relu(combined @ Wh1 + bh1)   [16x512]x[512x256] ------------
  {
    CDFrag acc[2];
#pragma unroll
    for (int t = 0; t < 2; ++t) acc[t].v = v8f{0, 0, 0, 0, 0, 0, 0, 0};
    gemm_rows16<2>(comb, 512, 0, p.wh1t, 512, wav * 2, 16, acc);
    epilogue_f32<2>(hs, H_DIM, 0, wav * 2, acc, p.bh1, true);
  }
  __syncthreads();

  // ---- P6b: q_value = h @ Wh2 + bh2 ---------------------------------------
  if (tid < BT) {
    const int r = tid;
    float acc = p.bh2[0];
    for (int n = 0; n < H_DIM; ++n) acc += hs[r * H_DIM + n] * p.wh2[n];
    p.out[b0 + r] = acc;
  }
}

// ---------------------------------------------------------------------------
extern "C" void kernel_launch(void* const* d_in, const int* in_sizes, int n_in,
                              void* d_out, int out_size, void* d_ws, size_t ws_size,
                              hipStream_t stream) {
  const float* obs   = (const float*)d_in[0];
  const float* skill = (const float*)d_in[1];
  const float* prm   = (const float*)d_in[2];
  const float* We1 = (const float*)d_in[3];
  const float* be1 = (const float*)d_in[4];
  const float* We2 = (const float*)d_in[5];
  const float* be2 = (const float*)d_in[6];
  const float* Wq  = (const float*)d_in[7];
  const float* bq  = (const float*)d_in[8];
  const float* Wk  = (const float*)d_in[9];
  const float* bk  = (const float*)d_in[10];
  const float* Wv  = (const float*)d_in[11];
  const float* bv  = (const float*)d_in[12];
  const float* Wo  = (const float*)d_in[13];
  const float* bo  = (const float*)d_in[14];
  const float* lng = (const float*)d_in[15];
  const float* lnb = (const float*)d_in[16];
  const float* Wh1 = (const float*)d_in[17];
  const float* bh1 = (const float*)d_in[18];
  const float* Wh2 = (const float*)d_in[19];
  const float* bh2 = (const float*)d_in[20];
  const int*   agent_idx = (const int*)d_in[21];

  unsigned short* ws = (unsigned short*)d_ws;
  const int B = in_sizes[0] / (A_N * OBS_D);   // 32768

  ac_prep_weights<<<512, 256, 0, stream>>>(We1, We2, Wq, Wk, Wv, Wo, Wh1, ws);

  FArgs a;
  a.obs = obs; a.skill = skill; a.prm = prm;
  a.we1t = ws + OFF_WE1T; a.we2t = ws + OFF_WE2T;
  a.wqt  = ws + OFF_WQT;  a.wkt  = ws + OFF_WKT;
  a.wvt  = ws + OFF_WVT;  a.wot  = ws + OFF_WOT;
  a.wh1t = ws + OFF_WH1T;
  a.be1 = be1; a.be2 = be2; a.bq = bq; a.bk = bk; a.bv = bv; a.bo = bo;
  a.lng = lng; a.lnb = lnb; a.bh1 = bh1; a.wh2 = Wh2; a.bh2 = bh2;
  a.agent_idx = agent_idx;
  a.out = (float*)d_out;
  a.B = B;

  ac_fused<<<B / BT, 256, 0, stream>>>(a);
}
